// MyModule_31653908971624
// MI455X (gfx1250) — matrix operand order repaired and checked by reading, not compiled, not run on hardware
//
#include <hip/hip_runtime.h>
#include <stdint.h>
#include <stddef.h>

// ---------------------------------------------------------------------------
// CDNA5 (gfx1250) wave32 WMMA types
// ---------------------------------------------------------------------------
typedef __attribute__((ext_vector_type(16))) __bf16        v16bf;
typedef __attribute__((ext_vector_type(8)))  float         v8f;
typedef __attribute__((ext_vector_type(4)))  unsigned int  u32x4;

__device__ __forceinline__ unsigned short f32_to_bf16_rne(float f) {
  union { float f; unsigned u; } c; c.f = f;
  unsigned u = c.u;
  u += 0x7fffu + ((u >> 16) & 1u);   // round-to-nearest-even
  return (unsigned short)(u >> 16);
}

// ---------------------------------------------------------------------------
// Layout transforms
// ---------------------------------------------------------------------------

// NCHW f32 -> NHWC bf16 (with total-channel stride + channel offset, for concat)
__global__ __launch_bounds__(256) void k_nchw_to_nhwc_bf16(
    const float* __restrict__ src, unsigned short* __restrict__ dst,
    int Bn, int Cs, int H, int W, int Ctot, int cOff)
{
  size_t n = (size_t)Bn * Cs * H * W;
  for (size_t i = (size_t)blockIdx.x * blockDim.x + threadIdx.x; i < n;
       i += (size_t)gridDim.x * blockDim.x) {
    int c = (int)(i % Cs); size_t r = i / Cs;
    int x = (int)(r % W); r /= W;
    int y = (int)(r % H); int b = (int)(r / H);
    float v = src[(((size_t)b * Cs + c) * H + y) * W + x];
    dst[(((size_t)b * H + y) * W + x) * Ctot + cOff + c] = f32_to_bf16_rne(v);
  }
}

// w[Co][Ci][ks][ks] f32 -> wt[taps][CoPad][Ci] bf16 (zero-fill padded co rows)
__global__ __launch_bounds__(256) void k_weight_transform(
    const float* __restrict__ w, unsigned short* __restrict__ wt,
    int Co, int CoPad, int Ci, int taps)
{
  size_t n = (size_t)taps * CoPad * Ci;
  for (size_t i = (size_t)blockIdx.x * blockDim.x + threadIdx.x; i < n;
       i += (size_t)gridDim.x * blockDim.x) {
    int ci = (int)(i % Ci); size_t r = i / Ci;
    int co = (int)(r % CoPad); int tap = (int)(r / CoPad);
    unsigned short v = 0;
    if (co < Co) v = f32_to_bf16_rne(w[((size_t)co * Ci + ci) * taps + tap]);
    wt[i] = v;
  }
}

// ---------------------------------------------------------------------------
// LDS-staged implicit-GEMM conv (KS=3 SAME or KS=1), v_wmma_f32_16x16x32_bf16.
//
// Block = 8 waves sharing (b = blockIdx.z, y = blockIdx.y); each wave owns one
// co-tile of 16 (coT = blockIdx.x*8 + waveId) and 3 accumulators covering the
// full W=48 output row.  For each (ky, ci0-chunk of 32) the input row segment
// [x=-1..48] x 32ch is staged into LDS via async global->LDS copies
// (double-buffered, s_wait_asynccnt), halo pre-zeroed so the WMMA consume path
// is branch-free (EXEC all ones).  Pixel stride 40 bf16 (80B) => conflict-free
// ds_load_b128 B-fragments across all 32 lanes.
//
// Fragment layouts per CDNA5 ISA 7.12.2 (wave32, 16-bit):
//   A 16x32: lane m=lane&15; half=lane>>4 holds K [8h..8h+7] and [16+8h..+7]
//   B 32x16: lane n=lane&15; half holds K [8h..8h+7] and [16+8h..+7]
//   D 16x16: lane n=lane&15; VGPR v -> m = v + 8*half
// ---------------------------------------------------------------------------
#define CONV_PSTR 40   // bf16 elements per staged pixel (80 bytes)
#define CONV_PIX  50   // staged pixels: x = -1 .. 48

template<int KS>
__global__ __launch_bounds__(256) void conv_wmma_lds_kernel(
    const unsigned short* __restrict__ wt,   // [KS*KS][CoPad][Ci] bf16
    const unsigned short* __restrict__ in,   // [B][H][W][Ci]      bf16 (NHWC)
    const float* __restrict__ bias,          // [Co] or nullptr
    float* __restrict__ out,                 // [B][Co][H][W]      f32 (NCHW)
    int Bn, int H, int W, int Ci, int Co, int CoPad)
{
  __shared__ __align__(16) unsigned short sh[2][CONV_PIX * CONV_PSTR];

  const int widx = threadIdx.x >> 5;
  const int coT  = blockIdx.x * 8 + widx;
  const int y    = blockIdx.y;
  const int b    = blockIdx.z;
  const int lane = threadIdx.x & 31;
  const int half = lane >> 4;
  const int lm   = lane & 15;
  const int co0  = coT << 4;

  // zero both buffers once (keeps halo pixels 0 and 49 at zero forever;
  // async copies only ever write pixels 1..48, bytes 0..63 of each slot)
  for (int i = threadIdx.x; i < 2 * CONV_PIX * CONV_PSTR; i += 256)
    ((unsigned short*)sh)[i] = 0;
  __syncthreads();

  const unsigned shBase = (unsigned)(uintptr_t)(&sh[0][0]);

  int kyArr[KS]; int nKy = 0;
#pragma unroll
  for (int t = 0; t < KS; ++t) {
    int ky = (KS == 3) ? (t - 1) : 0;
    int yy = y + ky;
    if (yy >= 0 && yy < H) kyArr[nKy++] = ky;
  }
  const int nCk   = Ci >> 5;
  const int total = nKy * nCk;

  const int tid = threadIdx.x;
  const int p   = tid >> 2;        // staged pixel 0..49 (first 200 threads)
  const int sub = tid & 3;         // 8-channel subchunk
  const int xx  = p - 1;
  const bool doCopy = (tid < 200) && (xx >= 0) && (xx < W);

  auto issue = [&](int ky, int ci0, int buf) {
    if (doCopy) {
      unsigned lds = shBase +
          (unsigned)((buf * CONV_PIX * CONV_PSTR + p * CONV_PSTR + sub * 8) * 2);
      const unsigned short* g =
          in + ((size_t)(b * H + (y + ky)) * W + xx) * Ci + ci0 + sub * 8;
      unsigned long long ga = (unsigned long long)(uintptr_t)g;
      asm volatile("global_load_async_to_lds_b128 %0, %1, off"
                   :: "v"(lds), "v"(ga) : "memory");
    }
  };

  v8f acc[3] = {};

  int iky = 0, ici = 0;            // issue cursor
  issue(kyArr[0], 0, 0);
  if (++ici == nCk) { ici = 0; ++iky; }
  int pky = 0, pci = 0;            // process cursor

  for (int c = 0; c < total; ++c) {
    if (c + 1 < total) {
      issue(kyArr[iky], ici << 5, (c + 1) & 1);
      if (++ici == nCk) { ici = 0; ++iky; }
      asm volatile("s_wait_asynccnt 0x1" ::: "memory");  // chunk c landed
    } else {
      asm volatile("s_wait_asynccnt 0x0" ::: "memory");
    }
    __syncthreads();

    const int ky  = kyArr[pky];
    const int ci0 = pci << 5;
    if (++pci == nCk) { pci = 0; ++pky; }
    const unsigned short* shBuf = &sh[c & 1][0];

#pragma unroll
    for (int kxi = 0; kxi < KS; ++kxi) {
      const int kx  = (KS == 3) ? (kxi - 1) : 0;
      const int tap = (KS == 3) ? ((ky + 1) * 3 + kxi) : 0;
      const unsigned short* wrow =
          wt + ((size_t)tap * CoPad + (co0 + lm)) * Ci + ci0 + half * 8;
      union { v16bf v; u32x4 q[2]; } am;
      am.q[0] = *(const u32x4*)(wrow);        // K 8h..8h+7
      am.q[1] = *(const u32x4*)(wrow + 16);   // K 16+8h..+7
#pragma unroll
      for (int t = 0; t < 3; ++t) {
        const int pp = t * 16 + lm + kx + 1;  // staged pixel (halo handles edges)
        const unsigned short* bp = shBuf + pp * CONV_PSTR + half * 8;
        union { v16bf v; u32x4 q[2]; } bm;
        bm.q[0] = *(const u32x4*)(bp);        // K 8h .. 8h+7 (ds_load_b128)
        bm.q[1] = *(const u32x4*)(bp + 16);   // K 16+8h .. 16+8h+7
        acc[t] = __builtin_amdgcn_wmma_f32_16x16x32_bf16(
            false, am.v, false, bm.v, (short)0, acc[t], false, false);
      }
    }
    __syncthreads();   // next iteration's issue overwrites this buffer
  }

#pragma unroll
  for (int t = 0; t < 3; ++t) {
    const int xo = t * 16 + lm;
#pragma unroll
    for (int v = 0; v < 8; ++v) {
      const int co = co0 + v + half * 8;
      if (co < Co) {
        float r = acc[t][v] + (bias ? bias[co] : 0.0f);
        out[(((size_t)b * Co + co) * H + y) * W + xo] = r;
      }
    }
  }
}

// ---------------------------------------------------------------------------
// Direct 1x1 conv for tiny Co (CoPad==32, nCo==2): q/k projections, classifier.
// Block = 8 waves: waveId -> (coT = w&1, ySub = w>>1); grid (H/4, B).
// No divisions, no predication in the WMMA path.
// ---------------------------------------------------------------------------
__global__ __launch_bounds__(256) void conv1x1_wmma_direct(
    const unsigned short* __restrict__ wt,   // [32][Ci] bf16
    const unsigned short* __restrict__ in,   // [B][H][W][Ci] bf16
    const float* __restrict__ bias,          // [Co] or nullptr
    float* __restrict__ out,                 // [B][Co][H][W] f32
    int Bn, int H, int W, int Ci, int Co, int CoPad)
{
  const int widx = threadIdx.x >> 5;
  const int coT  = widx & 1;
  const int y    = blockIdx.x * 4 + (widx >> 1);
  const int b    = blockIdx.y;
  const int lane = threadIdx.x & 31;
  const int half = lane >> 4;
  const int lm   = lane & 15;
  const int co0  = coT << 4;

  v8f acc[3] = {};
  const unsigned short* wrow = wt + (size_t)(co0 + lm) * Ci + half * 8;
  const unsigned short* irow = in + (size_t)(b * H + y) * W * Ci + half * 8;
  for (int ci0 = 0; ci0 < Ci; ci0 += 32) {
    union { v16bf v; u32x4 q[2]; } am;
    am.q[0] = *(const u32x4*)(wrow + ci0);
    am.q[1] = *(const u32x4*)(wrow + ci0 + 16);
    if (ci0 + 32 < Ci) __builtin_prefetch(wrow + ci0 + 32, 0, 3);
#pragma unroll
    for (int t = 0; t < 3; ++t) {
      const unsigned short* bp = irow + (size_t)(t * 16 + lm) * Ci + ci0;
      union { v16bf v; u32x4 q[2]; } bm;
      bm.q[0] = *(const u32x4*)(bp);
      bm.q[1] = *(const u32x4*)(bp + 16);
      acc[t] = __builtin_amdgcn_wmma_f32_16x16x32_bf16(
          false, am.v, false, bm.v, (short)0, acc[t], false, false);
    }
  }
#pragma unroll
  for (int t = 0; t < 3; ++t) {
    const int xo = t * 16 + lm;
#pragma unroll
    for (int v = 0; v < 8; ++v) {
      const int co = co0 + v + half * 8;
      if (co < Co) {
        float r = acc[t][v] + (bias ? bias[co] : 0.0f);
        out[(((size_t)b * Co + co) * H + y) * W + xo] = r;
      }
    }
  }
}

// ---------------------------------------------------------------------------
// BatchNorm (training stats) — one block per channel, then fused apply+ReLU
// ---------------------------------------------------------------------------
__global__ __launch_bounds__(256) void k_bn_stats(
    const float* __restrict__ x, float* __restrict__ mean, float* __restrict__ rinv,
    int Bn, int C, int HW)
{
  const int c = blockIdx.x;
  float s = 0.f, s2 = 0.f;
  for (int b = 0; b < Bn; ++b) {
    const float* p = x + ((size_t)b * C + c) * HW;
    for (int i = threadIdx.x; i < HW; i += blockDim.x) {
      float v = p[i]; s += v; s2 += v * v;
    }
  }
  __shared__ float sh0[256], sh1[256];
  sh0[threadIdx.x] = s; sh1[threadIdx.x] = s2;
  __syncthreads();
  for (int o = 128; o > 0; o >>= 1) {
    if (threadIdx.x < o) { sh0[threadIdx.x] += sh0[threadIdx.x + o]; sh1[threadIdx.x] += sh1[threadIdx.x + o]; }
    __syncthreads();
  }
  if (threadIdx.x == 0) {
    float n = (float)Bn * (float)HW;
    float m = sh0[0] / n;
    float var = fmaxf(sh1[0] / n - m * m, 0.0f);
    mean[c] = m;
    rinv[c] = rsqrtf(var + 1e-5f);
  }
}

__global__ __launch_bounds__(256) void k_bn_apply(
    const float* __restrict__ x, const float* __restrict__ g, const float* __restrict__ bt,
    const float* __restrict__ mean, const float* __restrict__ rinv,
    float* __restrict__ out_nchw,            // nullable
    unsigned short* __restrict__ out_nhwc,   // nullable (bf16, Ctot stride + offset)
    int Bn, int C, int H, int W, int Ctot, int cOff)
{
  size_t n = (size_t)Bn * C * H * W;
  for (size_t i = (size_t)blockIdx.x * blockDim.x + threadIdx.x; i < n;
       i += (size_t)gridDim.x * blockDim.x) {
    int xw = (int)(i % W); size_t r = i / W;
    int y = (int)(r % H); r /= H;
    int c = (int)(r % C); int b = (int)(r / C);
    float v = (x[i] - mean[c]) * rinv[c] * g[c] + bt[c];
    v = fmaxf(v, 0.0f);
    if (out_nchw) out_nchw[i] = v;
    if (out_nhwc)
      out_nhwc[(((size_t)b * H + y) * W + xw) * Ctot + cOff + c] = f32_to_bf16_rne(v);
  }
}

// ---------------------------------------------------------------------------
// Criss-cross attention (cc path).  q,k: [B][32][H][W]; v: [B][256][H][W].
// att[b,h,w, 0..H-1]=eH (diag -inf), [H..H+W-1]=eW; softmax over H+W=96.
// ---------------------------------------------------------------------------
__global__ __launch_bounds__(256) void k_cc_energy(
    const float* __restrict__ q, const float* __restrict__ k,
    float* __restrict__ att, int Bn, int Cq, int H, int W)
{
  const int L = H + W;
  const size_t HWs = (size_t)H * W;
  size_t n = (size_t)Bn * H * W * L;
  for (size_t i = (size_t)blockIdx.x * blockDim.x + threadIdx.x; i < n;
       i += (size_t)gridDim.x * blockDim.x) {
    int j = (int)(i % L); size_t r = i / L;
    int w = (int)(r % W); r /= W;
    int h = (int)(r % H); int b = (int)(r / H);
    const size_t qbase = (size_t)b * Cq * HWs + (size_t)h * W + w;
    float s;
    if (j < H) {
      if (j == h) {
        s = -__builtin_inff();
      } else {
        const size_t kbase = (size_t)b * Cq * HWs + (size_t)j * W + w;
        s = 0.f;
#pragma unroll 8
        for (int c = 0; c < Cq; ++c) s += q[qbase + c * HWs] * k[kbase + c * HWs];
      }
    } else {
      const int vv = j - H;
      const size_t kbase = (size_t)b * Cq * HWs + (size_t)h * W + vv;
      s = 0.f;
#pragma unroll 8
      for (int c = 0; c < Cq; ++c) s += q[qbase + c * HWs] * k[kbase + c * HWs];
    }
    att[i] = s;
  }
}

__global__ __launch_bounds__(256) void k_softmax_rows(
    float* __restrict__ att, int rows, int L)
{
  int r = blockIdx.x * blockDim.x + threadIdx.x;
  if (r >= rows) return;
  float* p = att + (size_t)r * L;
  float mx = -__builtin_inff();
  for (int l = 0; l < L; ++l) mx = fmaxf(mx, p[l]);
  float s = 0.f;
  for (int l = 0; l < L; ++l) { float e = __expf(p[l] - mx); p[l] = e; s += e; }
  float inv = 1.0f / s;
  for (int l = 0; l < L; ++l) p[l] *= inv;
}

__global__ __launch_bounds__(256) void k_cc_apply(
    const float* __restrict__ v, const float* __restrict__ att,
    const float* __restrict__ prev, const float* __restrict__ gamma,
    float* __restrict__ out_nchw, unsigned short* __restrict__ out_nhwc,
    int Bn, int C, int H, int W)
{
  const float gm = gamma[0];
  size_t n = (size_t)Bn * C * H * W;
  for (size_t i = (size_t)blockIdx.x * blockDim.x + threadIdx.x; i < n;
       i += (size_t)gridDim.x * blockDim.x) {
    int x = (int)(i % W); size_t r = i / W;
    int h = (int)(r % H); r /= H;
    int c = (int)(r % C); int b = (int)(r / C);
    const float* a  = att + (((size_t)b * H + h) * W + x) * (H + W);
    const float* vb = v + ((size_t)b * C + c) * H * W;
    float s = 0.f;
    for (int g = 0; g < H; ++g)  s += vb[(size_t)g * W + x] * a[g];          // oH
    const float* vr = vb + (size_t)h * W;
    for (int w2 = 0; w2 < W; ++w2) s += vr[w2] * a[H + w2];                  // oW
    float rr = gm * s + prev[i];
    out_nchw[i] = rr;
    out_nhwc[(((size_t)b * H + h) * W + x) * C + c] = f32_to_bf16_rne(rr);
  }
}

// ---------------------------------------------------------------------------
// "my" grid attention: to_grid => channel=(c,h,w), 4x4 grid = batch index.
// 16x16 batch Gram + per-batch softmax over 8 neighbors + 8-plane combine.
// ---------------------------------------------------------------------------
__global__ __launch_bounds__(256) void k_gram(
    const float* __restrict__ q, const float* __restrict__ k,
    float* __restrict__ gram, int n)
{
  const int b1 = blockIdx.x >> 4, b2 = blockIdx.x & 15;
  const float* qp = q + (size_t)b1 * n;
  const float* kp = k + (size_t)b2 * n;
  float s = 0.f;
  for (int i = threadIdx.x; i < n; i += blockDim.x) s += qp[i] * kp[i];
  __shared__ float sh[256];
  sh[threadIdx.x] = s; __syncthreads();
  for (int o = 128; o > 0; o >>= 1) {
    if (threadIdx.x < o) sh[threadIdx.x] += sh[threadIdx.x + o];
    __syncthreads();
  }
  if (threadIdx.x == 0) gram[blockIdx.x] = sh[0];
}

__global__ void k_my_coef(const float* __restrict__ gram, float* __restrict__ coef)
{
  int b1 = threadIdx.x;
  if (b1 >= 16) return;
  int gy = b1 >> 2, gx = b1 & 3;
  float e[8];
#pragma unroll
  for (int g = 0; g < 4; ++g)
    e[g] = (g == gy) ? -__builtin_inff() : gram[b1 * 16 + (g * 4 + gx)];   // eH, diag -inf
#pragma unroll
  for (int vv = 0; vv < 4; ++vv)
    e[4 + vv] = gram[b1 * 16 + (gy * 4 + vv)];                             // eW
  float mx = e[0];
  for (int j = 1; j < 8; ++j) mx = fmaxf(mx, e[j]);
  float s = 0.f, ex[8];
  for (int j = 0; j < 8; ++j) { ex[j] = __expf(e[j] - mx); s += ex[j]; }
  float inv = 1.0f / s;
  for (int j = 0; j < 8; ++j) coef[b1 * 8 + j] = ex[j] * inv;
}

__global__ __launch_bounds__(256) void k_my_apply(
    const float* __restrict__ v, const float* __restrict__ coef,
    const float* __restrict__ prev, const float* __restrict__ gamma,
    float* __restrict__ out_nchw, unsigned short* __restrict__ out_nhwc,
    int C, int H, int W)
{
  const size_t plane = (size_t)C * H * W;
  const size_t n = 16 * plane;
  const float gm = gamma[0];
  for (size_t i = (size_t)blockIdx.x * blockDim.x + threadIdx.x; i < n;
       i += (size_t)gridDim.x * blockDim.x) {
    size_t cp = i % plane;
    int b1 = (int)(i / plane);
    int gy = b1 >> 2, gx = b1 & 3;
    const float* cf = coef + b1 * 8;
    float s = 0.f;
#pragma unroll
    for (int g = 0; g < 4; ++g)   s += cf[g]      * v[(size_t)(g * 4 + gx) * plane + cp];
#pragma unroll
    for (int vv = 0; vv < 4; ++vv) s += cf[4 + vv] * v[(size_t)(gy * 4 + vv) * plane + cp];
    float r = gm * s + prev[i];
    out_nchw[i] = r;
    int x = (int)(cp % W); size_t t = cp / W;
    int y = (int)(t % H); int c = (int)(t / H);
    out_nhwc[(((size_t)b1 * H + y) * W + x) * C + c] = f32_to_bf16_rne(r);
  }
}

// ---------------------------------------------------------------------------
// Orchestration
// ---------------------------------------------------------------------------
extern "C" void kernel_launch(void* const* d_in, const int* in_sizes, int n_in,
                              void* d_out, int out_size, void* d_ws, size_t ws_size,
                              hipStream_t stream)
{
  (void)in_sizes; (void)n_in; (void)out_size; (void)ws_size;

  const float* x        = (const float*)d_in[0];
  const float* w_a      = (const float*)d_in[1];
  const float* g_a      = (const float*)d_in[2];
  const float* b_a      = (const float*)d_in[3];
  const float* ccq_w    = (const float*)d_in[4];
  const float* ccq_b    = (const float*)d_in[5];
  const float* cck_w    = (const float*)d_in[6];
  const float* cck_b    = (const float*)d_in[7];
  const float* ccv_w    = (const float*)d_in[8];
  const float* ccv_b    = (const float*)d_in[9];
  const float* cc_gamma = (const float*)d_in[10];
  const float* w_b      = (const float*)d_in[11];
  const float* g_b      = (const float*)d_in[12];
  const float* b_bb     = (const float*)d_in[13];
  const float* w_m1     = (const float*)d_in[14];
  const float* g_m1     = (const float*)d_in[15];
  const float* b_m1     = (const float*)d_in[16];
  const float* myq_w    = (const float*)d_in[17];
  const float* myq_b    = (const float*)d_in[18];
  const float* myk_w    = (const float*)d_in[19];
  const float* myk_b    = (const float*)d_in[20];
  const float* myv_w    = (const float*)d_in[21];
  const float* myv_b    = (const float*)d_in[22];
  const float* my_gamma = (const float*)d_in[23];
  const float* w_m2     = (const float*)d_in[24];
  const float* g_m2     = (const float*)d_in[25];
  const float* b_m2     = (const float*)d_in[26];
  const float* w_c      = (const float*)d_in[27];
  const float* g_c      = (const float*)d_in[28];
  const float* b_c      = (const float*)d_in[29];
  const float* w_cls    = (const float*)d_in[30];
  const float* b_cls    = (const float*)d_in[31];

  const int Bn = 16, Cin = 1024, H = 48, W = 48;
  const int HW = H * W;            // 2304
  const int Npix = Bn * HW;        // 36864
  const int C = 256, C8 = 32, OC = 512, NC = 21, CCAT = Cin + 2 * C; // 1536

  // ---- workspace plan (bump allocator, lifetime-based aliasing) ----
  char* ws = (char*)d_ws;
  size_t off = 0;
  auto alloc = [&](size_t bytes) -> char* {
    char* p = ws + off;
    off = (off + bytes + 255) & ~(size_t)255;
    return p;
  };
  char* regA  = alloc((size_t)Npix * CCAT * 2);      // xnhwc (ph0-1) / concat (ph4-5)
  char* regB  = alloc((size_t)Bn * OC * HW * 4);     // pre-BN conv scratch / h pre-BN
  float* conv1 = (float*)regB;
  float* conv2 = (float*)(regB + (size_t)Bn * C * HW * 4);
  float* conv3 = (float*)regB;
  char* ccAB0 = alloc((size_t)Bn * C * HW * 4);      // cc ping  / h_nhwc overlay (ph5)
  char* ccAB1 = alloc((size_t)Bn * C * HW * 4);      // cc pong
  char* myAB0 = alloc((size_t)Bn * C * HW * 4);
  char* myAB1 = alloc((size_t)Bn * C * HW * 4);
  unsigned short* ccNH = (unsigned short*)alloc((size_t)Npix * C * 2);
  unsigned short* myNH = (unsigned short*)alloc((size_t)Npix * C * 2);
  float* qbuf = (float*)alloc((size_t)Bn * C8 * HW * 4);
  float* kbuf = (float*)alloc((size_t)Bn * C8 * HW * 4);
  float* vbuf = (float*)alloc((size_t)Bn * C  * HW * 4);
  float* attb = (float*)alloc((size_t)Npix * (H + W) * 4);
  unsigned short* wtA   = (unsigned short*)alloc((size_t)9 * C * Cin * 2);
  unsigned short* wtM1  = (unsigned short*)alloc((size_t)9 * C * Cin * 2);
  unsigned short* wtWB  = (unsigned short*)alloc((size_t)9 * C * C * 2);
  unsigned short* wtWM2 = (unsigned short*)alloc((size_t)9 * C * C * 2);
  unsigned short* wtWC  = (unsigned short*)alloc((size_t)9 * OC * CCAT * 2);
  unsigned short* wtQ   = (unsigned short*)alloc((size_t)C8 * C * 2);
  unsigned short* wtK   = (unsigned short*)alloc((size_t)C8 * C * 2);
  unsigned short* wtV   = (unsigned short*)alloc((size_t)C * C * 2);
  unsigned short* wtMQ  = (unsigned short*)alloc((size_t)C8 * C * 2);
  unsigned short* wtMK  = (unsigned short*)alloc((size_t)C8 * C * 2);
  unsigned short* wtMV  = (unsigned short*)alloc((size_t)C * C * 2);
  unsigned short* wtCLS = (unsigned short*)alloc((size_t)32 * OC * 2);
  float* meanb = (float*)alloc(512 * 4);
  float* rinvb = (float*)alloc(512 * 4);
  float* gramb = (float*)alloc(256 * 4);
  float* coefb = (float*)alloc(16 * 8 * 4);

  unsigned short* xnhwc  = (unsigned short*)regA;   // stride Cin
  unsigned short* concat = (unsigned short*)regA;   // stride CCAT
  unsigned short* hNH    = (unsigned short*)ccAB0;  // [Npix][OC]

  auto gsBlocks = [](size_t n) -> int {
    size_t b = (n + 255) / 256;
    return (int)(b > 1048576 ? 1048576 : b);
  };
  auto wtrans = [&](const float* w, unsigned short* wt, int Co, int CoPad, int Ci, int taps) {
    size_t n = (size_t)taps * CoPad * Ci;
    k_weight_transform<<<gsBlocks(n), 256, 0, stream>>>(w, wt, Co, CoPad, Ci, taps);
  };
  // LDS-staged conv: requires Co multiple of 128 (nCo multiple of 8)
  auto convLds3 = [&](const unsigned short* wt, const unsigned short* in, const float* bias,
                      float* out, int Ci, int Co) {
    dim3 g((Co / 16) / 8, H, Bn);
    conv_wmma_lds_kernel<3><<<g, 256, 0, stream>>>(wt, in, bias, out, Bn, H, W, Ci, Co, Co);
  };
  auto convLds1 = [&](const unsigned short* wt, const unsigned short* in, const float* bias,
                      float* out, int Ci, int Co) {
    dim3 g((Co / 16) / 8, H, Bn);
    conv_wmma_lds_kernel<1><<<g, 256, 0, stream>>>(wt, in, bias, out, Bn, H, W, Ci, Co, Co);
  };
  // direct small-Co 1x1 (CoPad = 32)
  auto convSmall = [&](const unsigned short* wt, const unsigned short* in, const float* bias,
                       float* out, int Ci, int Co) {
    dim3 g(H / 4, Bn);
    conv1x1_wmma_direct<<<g, 256, 0, stream>>>(wt, in, bias, out, Bn, H, W, Ci, Co, 32);
  };
  auto bnStatsApply = [&](float* pre, const float* g, const float* bt, int Cc,
                          float* out_nchw, unsigned short* out_nhwc, int Ctot, int cOff) {
    k_bn_stats<<<Cc, 256, 0, stream>>>(pre, meanb, rinvb, Bn, Cc, HW);
    size_t n = (size_t)Bn * Cc * HW;
    k_bn_apply<<<gsBlocks(n), 256, 0, stream>>>(pre, g, bt, meanb, rinvb,
                                                out_nchw, out_nhwc, Bn, Cc, H, W, Ctot, cOff);
  };

  // ---- Phase 0: layout transforms (bf16 NHWC activations, bf16 weights) ----
  { size_t n = (size_t)Bn * Cin * HW;
    k_nchw_to_nhwc_bf16<<<gsBlocks(n), 256, 0, stream>>>(x, xnhwc, Bn, Cin, H, W, Cin, 0); }
  wtrans(w_a,   wtA,   C,  C,  Cin,  9);
  wtrans(w_m1,  wtM1,  C,  C,  Cin,  9);
  wtrans(w_b,   wtWB,  C,  C,  C,    9);
  wtrans(w_m2,  wtWM2, C,  C,  C,    9);
  wtrans(w_c,   wtWC,  OC, OC, CCAT, 9);
  wtrans(ccq_w, wtQ,   C8, 32, C,    1);
  wtrans(cck_w, wtK,   C8, 32, C,    1);
  wtrans(ccv_w, wtV,   C,  C,  C,    1);
  wtrans(myq_w, wtMQ,  C8, 32, C,    1);
  wtrans(myk_w, wtMK,  C8, 32, C,    1);
  wtrans(myv_w, wtMV,  C,  C,  C,    1);
  wtrans(w_cls, wtCLS, NC, 32, OC,   1);

  // ---- Phase 1: conv3x3(w_a)/(w_m1) + BN+ReLU ----
  convLds3(wtA,  xnhwc, nullptr, conv1, Cin, C);
  convLds3(wtM1, xnhwc, nullptr, conv2, Cin, C);
  bnStatsApply(conv1, g_a,  b_a,  C, (float*)ccAB0, ccNH, C, 0);
  bnStatsApply(conv2, g_m1, b_m1, C, (float*)myAB0, myNH, C, 0);

  // ---- Phase 2: criss-cross attention x2 ----
  float* ccP = (float*)ccAB0; float* ccN = (float*)ccAB1;
  for (int r = 0; r < 2; ++r) {
    convSmall(wtQ, ccNH, ccq_b, qbuf, C, C8);
    convSmall(wtK, ccNH, cck_b, kbuf, C, C8);
    convLds1(wtV, ccNH, ccv_b, vbuf, C, C);
    { size_t n = (size_t)Npix * (H + W);
      k_cc_energy<<<gsBlocks(n), 256, 0, stream>>>(qbuf, kbuf, attb, Bn, C8, H, W); }
    k_softmax_rows<<<(Npix + 255) / 256, 256, 0, stream>>>(attb, Npix, H + W);
    { size_t n = (size_t)Bn * C * HW;
      k_cc_apply<<<gsBlocks(n), 256, 0, stream>>>(vbuf, attb, ccP, cc_gamma, ccN, ccNH, Bn, C, H, W); }
    float* t = ccP; ccP = ccN; ccN = t;
  }

  // ---- Phase 3: grid ("my") attention x2 ----
  float* myP = (float*)myAB0; float* myN = (float*)myAB1;
  for (int r = 0; r < 2; ++r) {
    convSmall(wtMQ, myNH, myq_b, qbuf, C, C8);
    convSmall(wtMK, myNH, myk_b, kbuf, C, C8);
    convLds1(wtMV, myNH, myv_b, vbuf, C, C);
    k_gram<<<256, 256, 0, stream>>>(qbuf, kbuf, gramb, C8 * HW);
    k_my_coef<<<1, 32, 0, stream>>>(gramb, coefb);
    { size_t n = (size_t)Bn * C * HW;
      k_my_apply<<<gsBlocks(n), 256, 0, stream>>>(vbuf, coefb, myP, my_gamma, myN, myNH, C, H, W); }
    float* t = myP; myP = myN; myN = t;
  }

  // ---- Phase 4: conv3x3(w_b)/(w_m2) + BN into concat, plus x slice ----
  convLds3(wtWB, ccNH, nullptr, conv1, C, C);
  bnStatsApply(conv1, g_b, b_bb, C, nullptr, concat, CCAT, Cin);
  convLds3(wtWM2, myNH, nullptr, conv1, C, C);
  bnStatsApply(conv1, g_m2, b_m2, C, nullptr, concat, CCAT, Cin + C);
  { size_t n = (size_t)Bn * Cin * HW;
    k_nchw_to_nhwc_bf16<<<gsBlocks(n), 256, 0, stream>>>(x, concat, Bn, Cin, H, W, CCAT, 0); }

  // ---- Phase 5: conv3x3(w_c) + BN, then 1x1 classifier into d_out ----
  convLds3(wtWC, concat, nullptr, conv3, CCAT, OC);
  bnStatsApply(conv3, g_c, b_c, OC, nullptr, hNH, OC, 0);
  convSmall(wtCLS, hNH, b_cls, (float*)d_out, OC, NC);
}
